// MultiHeadAttentionBlock_13108240187636
// MI455X (gfx1250) — matrix operand order repaired and checked by reading, not compiled, not run on hardware
//
#include <hip/hip_runtime.h>
#include <hip/hip_bf16.h>

// ---------------------------------------------------------------------------
// MultiHeadAttentionBlock for MI455X (gfx1250): bf16 WMMA + double-buffered
// TDM pipeline.  B=16, C=D=256, H=W=32 -> S=1024, NH=8, DK=32, G=32 groups
// ---------------------------------------------------------------------------

#define BB   16
#define CC   256
#define DD   256
#define SS   1024           // H*W
#define NH   8
#define DK   32
#define NG   32
#define GSZ  8192           // (DD/NG)*SS elements per group
#define SCALE 0.1767766953f // 1/sqrt(32)
#define EPS  1e-5f

typedef __attribute__((ext_vector_type(16))) __bf16   v16bf;
typedef __attribute__((ext_vector_type(8)))  float    v8f;
typedef __attribute__((ext_vector_type(4)))  unsigned u32x4;
typedef __attribute__((ext_vector_type(8)))  int      i32x8;
typedef __attribute__((ext_vector_type(4)))  int      i32x4;

union ABpack { v16bf h; unsigned u[8]; };

// A-operand (16x32, 16-bit) K index for VGPR pair v (0..7), lane-half h
__device__ __forceinline__ int kA(int v, int h) {
    return 2 * (v & 3) + 16 * (v >> 2) + 8 * h;
}
// B-operand (32x16, 16-bit) K index
__device__ __forceinline__ int kB(int v, int h) {
    return 2 * (v & 3) + 16 * (v >> 2) + 8 * h;
}

__device__ __forceinline__ v8f wmma_bf16(v16bf a, v16bf b, v8f c) {
    return __builtin_amdgcn_wmma_f32_16x16x32_bf16(
        false, a, false, b, (short)0, c, false, false);
}

__device__ __forceinline__ float rowmax16(float x) {
#pragma unroll
    for (int m = 8; m; m >>= 1) x = fmaxf(x, __shfl_xor(x, m, 32));
    return x;
}
__device__ __forceinline__ float rowsum16(float x) {
#pragma unroll
    for (int m = 8; m; m >>= 1) x += __shfl_xor(x, m, 32);
    return x;
}

// ---------------------------------------------------------------------------
// Tensor Data Mover: 2-D tile load (global -> LDS).
//   D# group0: count=1 | lds_addr | global_addr | type=2
//   D# group1: data_size | tensor_dim0/1 | tile_dim0/1 | tensor_dim0_stride
// size_code: 0=1B, 1=2B (bf16), 2=4B (f32)
//
// The trailing asm is essential and must CAPTURE the LDS pointer: the
// compiler does not model the TDM builtin as writing LDS, and a bare
// memory-clobber asm cannot alias a non-escaped internal LDS global (round-3
// disasm: LDS loads still folded to undef).  Passing the destination pointer
// as an operand escapes it, so the asm may write it and the loads survive.
// ---------------------------------------------------------------------------
__device__ __forceinline__ void tdm_load_2d(
    void* lds_ptr, unsigned lds_off, const void* gp,
    unsigned tdim0, unsigned tdim1, unsigned stride0,
    unsigned tile0, unsigned tile1, unsigned size_code)
{
    unsigned long long ga = (unsigned long long)(uintptr_t)gp;
    u32x4 g0;
    g0[0] = 1u;                                   // count=1, user mode
    g0[1] = lds_off;                              // LDS byte address
    g0[2] = (unsigned)(ga & 0xFFFFFFFFu);         // global_addr[31:0]
    g0[3] = (unsigned)((ga >> 32) & 0x01FFFFFFu)  // global_addr[56:32]
          | (2u << 30);                           // type = 2 ("image")
    i32x8 g1;
    g1[0] = (int)(size_code << 16);               // data_size
    g1[1] = (int)((tdim0 & 0xFFFFu) << 16);       // tensor_dim0[15:0]
    g1[2] = (int)(((tdim0 >> 16) & 0xFFFFu) |     // tensor_dim0[31:16]
                  ((tdim1 & 0xFFFFu) << 16));     // tensor_dim1[15:0]
    g1[3] = (int)(((tdim1 >> 16) & 0xFFFFu) |     // tensor_dim1[31:16]
                  ((tile0 & 0xFFFFu) << 16));     // tile_dim0
    g1[4] = (int)(tile1 & 0xFFFFu);               // tile_dim1 (tile_dim2 = 0)
    g1[5] = (int)stride0;                         // tensor_dim0_stride[31:0]
    g1[6] = 0;                                    // stride hi / dim1_stride
    g1[7] = 0;
    i32x4 z = {0, 0, 0, 0};
#if defined(__clang_major__) && __clang_major__ >= 23
    i32x8 z8 = {0, 0, 0, 0, 0, 0, 0, 0};
    __builtin_amdgcn_tensor_load_to_lds(g0, g1, z, z, z8, 0);
#else
    __builtin_amdgcn_tensor_load_to_lds(g0, g1, z, z, 0);
#endif
    // Escape the LDS destination: TDM wrote it behind the compiler's back.
    asm volatile("" : : "v"(lds_ptr) : "memory");
}

// ---------------------------------------------------------------------------
// Kernel 1: QKV projection.  Out[b][s][d] = sum_c X[b][c][s] * W[c][d]
//   Block = 4 waves on one 64-wide d-block; wave w -> rows s0..s0+15.
//   Double-buffered TDM: stage (i+1) issued before computing stage i.
//   Per stage: W tile 32x64 f32 (wave0) + per-wave X tile 32x16 f32 (row
//   stride S).  4 wmma per wave per stage (A reused across 4 N-tiles).
// ---------------------------------------------------------------------------
__global__ __launch_bounds__(128) void proj_kernel(
    const float* __restrict__ X, const float* __restrict__ W,
    __bf16* __restrict__ Out)
{
    __shared__ char smem[32768];  // W bufs 2x8KB | A bufs 2x(4 waves x 2KB)

    const int lane = threadIdx.x & 31;
    const int wv   = threadIdx.x >> 5;
    const int l16  = lane & 15;
    const int half = lane >> 4;

    const int b   = blockIdx.x >> 6;                 // 64 blocks per batch
    const int rem = blockIdx.x & 63;
    const int s0  = ((rem >> 2) << 6) + (wv << 4);   // 16 s-groups of 64 rows
    const int d0  = (rem & 3) << 6;                  // 4 d-blocks of 64 cols

    const float* Xb = X + (size_t)b * CC * SS;

    auto issue_stage = [&](int i) {
        const int c0 = i * 32;
        const unsigned j = (unsigned)(i & 1);
        const unsigned offA = 16384u + j * 8192u + (unsigned)wv * 2048u;
        tdm_load_2d(smem + offA, offA,
                    &Xb[(size_t)c0 * SS + s0], 16, 32, SS, 16, 32, /*f32*/2);
        if (wv == 0) {
            const unsigned offW = j * 8192u;
            tdm_load_2d(smem + offW, offW, &W[(size_t)c0 * DD + d0],
                        64, 32, DD, 64, 32, /*f32*/2);
        }
    };

    v8f acc[4] = {};
    issue_stage(0);
#pragma unroll 1
    for (int i = 0; i < CC / 32; ++i) {
        if (i + 1 < CC / 32) {
            issue_stage(i + 1);                  // keep next stage in flight
            if (wv == 0) __builtin_amdgcn_s_wait_tensorcnt(2);
            else         __builtin_amdgcn_s_wait_tensorcnt(1);
        } else {
            __builtin_amdgcn_s_wait_tensorcnt(0);
        }
        __syncthreads();

        const float* lw = (const float*)(smem + (i & 1) * 8192);
        const float* la = (const float*)(smem + 16384 + (i & 1) * 8192
                                              + wv * 2048);
        v16bf a;
#pragma unroll
        for (int v = 0; v < 8; ++v) {
            const int ka = kA(v, half);
            a[2 * v]     = (__bf16)la[ka * 16 + l16];
            a[2 * v + 1] = (__bf16)la[(ka + 1) * 16 + l16];
        }
#pragma unroll
        for (int t = 0; t < 4; ++t) {
            v16bf bb;
#pragma unroll
            for (int v = 0; v < 8; ++v) {
                const int kb = kB(v, half);
                bb[2 * v]     = (__bf16)lw[kb * 64 + t * 16 + l16];
                bb[2 * v + 1] = (__bf16)lw[(kb + 1) * 64 + t * 16 + l16];
            }
            acc[t] = wmma_bf16(a, bb, acc[t]);
        }
        __syncthreads();   // all waves done with buf (i&1) before stage i+2
    }

    __bf16* O = Out + ((size_t)b * SS + s0) * DD + d0;
#pragma unroll
    for (int t = 0; t < 4; ++t)
#pragma unroll
        for (int v = 0; v < 8; ++v) {
            const int m = v + 8 * half;
            O[(size_t)m * DD + t * 16 + l16] = (__bf16)acc[t][v];
        }
}

// ---------------------------------------------------------------------------
// Kernel 2: flash attention.  Block = 4 waves on one (b, h, 64-row block);
//   wave w -> query rows s0+w*16.  Double-buffered TDM staging of K and V
//   tiles (32 keys x 32 dk bf16, row stride D).  QK^T: 2 wmma; online
//   softmax in registers; P transposed via per-wave LDS bounce; PV: 2 wmma.
// ---------------------------------------------------------------------------
__global__ __launch_bounds__(128) void attn_kernel(
    const __bf16* __restrict__ Qb, const __bf16* __restrict__ Kb,
    const __bf16* __restrict__ Vb, __bf16* __restrict__ Cx)
{
    __shared__ char smem[8192 + 4 * 1024];  // 2 x (lk 2KB | lv 2KB) | bounce

    const int lane = threadIdx.x & 31;
    const int wv   = threadIdx.x >> 5;
    const int l16  = lane & 15;
    const int half = lane >> 4;

    const int rb = blockIdx.x & 15;       // 16 row-blocks of 64
    const int bh = blockIdx.x >> 4;
    const int h  = bh & (NH - 1);
    const int b  = bh >> 3;
    const int s0 = (rb << 6) + (wv << 4);

    const __bf16* Qp = Qb + ((size_t)b * SS + s0) * DD + h * DK;
    const __bf16* Kp = Kb + (size_t)b * SS * DD + h * DK;
    const __bf16* Vp = Vb + (size_t)b * SS * DD + h * DK;

    __bf16* pb = (__bf16*)(smem + 8192 + wv * 1024);

    auto issue_kv = [&](int p) {
        const unsigned j = (unsigned)(p & 1);
        tdm_load_2d(smem + j * 4096u, j * 4096u,
                    Kp + (size_t)(p * 32) * DD, 32, 32, DD, 32, 32, /*bf16*/1);
        tdm_load_2d(smem + j * 4096u + 2048u, j * 4096u + 2048u,
                    Vp + (size_t)(p * 32) * DD, 32, 32, DD, 32, 32, /*bf16*/1);
    };

    // Q A-operand (16x32), packed 32-bit loads
    ABpack qa;
#pragma unroll
    for (int v = 0; v < 8; ++v)
        qa.u[v] = *(const unsigned*)(Qp + (size_t)l16 * DD + kA(v, half));

    v8f ctx0 = {}, ctx1 = {};
    float Mrun[8], Lrun[8];
#pragma unroll
    for (int v = 0; v < 8; ++v) { Mrun[v] = -1e30f; Lrun[v] = 0.0f; }

    if (wv == 0) issue_kv(0);
#pragma unroll 1
    for (int p = 0; p < SS / 32; ++p) {
        if (wv == 0) {
            if (p + 1 < SS / 32) {
                issue_kv(p + 1);                 // next stage in flight
                __builtin_amdgcn_s_wait_tensorcnt(2);
                if (p + 2 < SS / 32) {           // warm L2 two stages ahead
                    __builtin_prefetch(Kp + (size_t)((p + 2) * 32) * DD, 0, 0);
                    __builtin_prefetch(Vp + (size_t)((p + 2) * 32) * DD, 0, 0);
                }
            } else {
                __builtin_amdgcn_s_wait_tensorcnt(0);
            }
        }
        __syncthreads();

        const __bf16* lkh = (const __bf16*)(smem + (p & 1) * 4096);
        const __bf16* lvh = lkh + 1024;

        // ---- scores: two 16x16 tiles over the 32 staged keys ----
        v8f sc0 = {}, sc1 = {};
#pragma unroll
        for (int t = 0; t < 2; ++t) {
            ABpack kv;   // B elem (k,n) = K[key t*16+n][k]
#pragma unroll
            for (int v = 0; v < 8; ++v)
                kv.u[v] = *(const unsigned*)(lkh + (t * 16 + l16) * 32 + kB(v, half));
            if (t == 0) sc0 = wmma_bf16(qa.h, kv.h, sc0);
            else        sc1 = wmma_bf16(qa.h, kv.h, sc1);
        }

        // ---- online softmax per row (row m = v + 8*half) ----
#pragma unroll
        for (int v = 0; v < 8; ++v) {
            const float a0 = sc0[v] * SCALE;
            const float a1 = sc1[v] * SCALE;
            const float Mnew = fmaxf(Mrun[v], rowmax16(fmaxf(a0, a1)));
            const float f    = __expf(Mrun[v] - Mnew);
            const float p0   = __expf(a0 - Mnew);
            const float p1   = __expf(a1 - Mnew);
            Lrun[v] = Lrun[v] * f + rowsum16(p0 + p1);
            Mrun[v] = Mnew;
            ctx0[v] *= f;
            ctx1[v] *= f;
            sc0[v] = p0;
            sc1[v] = p1;
        }

        // ---- P: C-layout f32 -> A-layout bf16 via per-wave bounce ----
#pragma unroll
        for (int v = 0; v < 8; ++v) {
            const int m = v + 8 * half;
            pb[m * 32 + l16]      = (__bf16)sc0[v];
            pb[m * 32 + 16 + l16] = (__bf16)sc1[v];
        }
        __syncthreads();
        ABpack pa;
#pragma unroll
        for (int v = 0; v < 8; ++v)
            pa.u[v] = *(const unsigned*)(pb + l16 * 32 + kA(v, half));

        // ---- ctx += P(16x32) @ V(32x32): two N tiles of dk ----
#pragma unroll
        for (int t = 0; t < 2; ++t) {
            v16bf vvb;   // B elem (k,n) = V[key k][n]
#pragma unroll
            for (int v = 0; v < 8; ++v) {
                const int k = kB(v, half);
                vvb[2 * v]     = lvh[k * 32       + t * 16 + l16];
                vvb[2 * v + 1] = lvh[(k + 1) * 32 + t * 16 + l16];
            }
            if (t == 0) ctx0 = wmma_bf16(pa.h, vvb, ctx0);
            else        ctx1 = wmma_bf16(pa.h, vvb, ctx1);
        }
        __syncthreads();   // buf (p&1) free before stage p+2 overwrites it
    }

    // ---- normalize, store merged-head context (bf16 [B,S,D]) ----
    __bf16* Co = Cx + ((size_t)b * SS + s0) * DD + h * DK;
#pragma unroll
    for (int v = 0; v < 8; ++v) {
        const float r = 1.0f / Lrun[v];
        const int m = v + 8 * half;
        Co[(size_t)m * DD + l16]      = (__bf16)(ctx0[v] * r);
        Co[(size_t)m * DD + 16 + l16] = (__bf16)(ctx1[v] * r);
    }
}

// ---------------------------------------------------------------------------
// Kernel 3: out-projection + residual, Y[b][d][s] (f32 channel-major).
//   Double-buffered TDM staging of Wo tiles; ctx rows read with packed loads.
// ---------------------------------------------------------------------------
__global__ __launch_bounds__(128) void oproj_kernel(
    const __bf16* __restrict__ Cx, const float* __restrict__ Wo,
    const __bf16* __restrict__ Vb, float* __restrict__ Y)
{
    __shared__ char smem[16384];  // 2 x Wo tile (32x64 f32)

    const int lane = threadIdx.x & 31;
    const int wv   = threadIdx.x >> 5;
    const int l16  = lane & 15;
    const int half = lane >> 4;

    const int b   = blockIdx.x >> 6;
    const int rem = blockIdx.x & 63;
    const int s0  = ((rem >> 2) << 6) + (wv << 4);
    const int d0  = (rem & 3) << 6;

    const __bf16* Ab = Cx + (size_t)b * SS * DD;

    auto issue_w = [&](int i) {
        const unsigned off = (unsigned)(i & 1) * 8192u;
        tdm_load_2d(smem + off, off,
                    &Wo[(size_t)(i * 32) * DD + d0], 64, 32, DD, 64, 32, 2);
    };

    v8f acc[4] = {};
    if (wv == 0) issue_w(0);
#pragma unroll 1
    for (int i = 0; i < DD / 32; ++i) {
        if (wv == 0) {
            if (i + 1 < DD / 32) {
                issue_w(i + 1);
                __builtin_amdgcn_s_wait_tensorcnt(1);
            } else {
                __builtin_amdgcn_s_wait_tensorcnt(0);
            }
        }
        __syncthreads();

        const float* lw = (const float*)(smem + (i & 1) * 8192);
        ABpack a;   // ctx rows are contiguous bf16: packed loads
#pragma unroll
        for (int v = 0; v < 8; ++v)
            a.u[v] = *(const unsigned*)(Ab + (size_t)(s0 + l16) * DD
                                           + i * 32 + kA(v, half));
#pragma unroll
        for (int t = 0; t < 4; ++t) {
            v16bf bb;
#pragma unroll
            for (int v = 0; v < 8; ++v) {
                const int kb = kB(v, half);
                bb[2 * v]     = (__bf16)lw[kb * 64 + t * 16 + l16];
                bb[2 * v + 1] = (__bf16)lw[(kb + 1) * 64 + t * 16 + l16];
            }
            acc[t] = wmma_bf16(a.h, bb, acc[t]);
        }
        __syncthreads();
    }

#pragma unroll
    for (int t = 0; t < 4; ++t) {
        const int d = d0 + t * 16 + l16;
#pragma unroll
        for (int v = 0; v < 8; ++v) {
            const int m = v + 8 * half;
            const int s = s0 + m;
            const float res = (float)Vb[((size_t)b * SS + s) * DD + d];
            Y[((size_t)b * DD + d) * SS + s] = acc[t][v] + res;
        }
    }
}

// ---------------------------------------------------------------------------
// Kernel 4: fused GroupNorm (32 groups of 8 channels x 1024 pixels).
// ---------------------------------------------------------------------------
__global__ __launch_bounds__(256) void gn_kernel(
    const float* __restrict__ Y, const float* __restrict__ gamma,
    const float* __restrict__ beta, float* __restrict__ out)
{
    __shared__ float sred[512];

    const int b = blockIdx.x >> 5;
    const int g = blockIdx.x & 31;
    const float* base = Y + ((size_t)b * DD + g * 8) * SS;

    float sum = 0.f, sq = 0.f;
    for (int i = threadIdx.x; i < GSZ; i += 256) {
        const float x = base[i];
        sum += x;
        sq  += x * x;
    }
    sred[threadIdx.x]       = sum;
    sred[256 + threadIdx.x] = sq;
    __syncthreads();
#pragma unroll
    for (int st = 128; st; st >>= 1) {
        if (threadIdx.x < st) {
            sred[threadIdx.x]       += sred[threadIdx.x + st];
            sred[256 + threadIdx.x] += sred[256 + threadIdx.x + st];
        }
        __syncthreads();
    }
    const float mu   = sred[0] * (1.0f / GSZ);
    const float var  = sred[256] * (1.0f / GSZ) - mu * mu;
    const float rstd = rsqrtf(var + EPS);

    float* obase = out + ((size_t)b * DD + g * 8) * SS;
    for (int i = threadIdx.x; i < GSZ; i += 256) {
        const int d = g * 8 + (i >> 10);
        obase[i] = (base[i] - mu) * rstd * gamma[d] + beta[d];
    }
}

// ---------------------------------------------------------------------------
extern "C" void kernel_launch(void* const* d_in, const int* in_sizes, int n_in,
                              void* d_out, int out_size, void* d_ws, size_t ws_size,
                              hipStream_t stream)
{
    (void)in_sizes; (void)n_in; (void)out_size; (void)ws_size;

    const float* q     = (const float*)d_in[0];
    const float* k     = (const float*)d_in[1];
    const float* v     = (const float*)d_in[2];
    const float* Wq    = (const float*)d_in[3];
    const float* Wk    = (const float*)d_in[4];
    const float* Wv    = (const float*)d_in[5];
    const float* Wo    = (const float*)d_in[6];
    const float* gamma = (const float*)d_in[7];
    const float* beta  = (const float*)d_in[8];

    // workspace: Qb/Kb/Vb/Cx bf16 [B,S,D] (8 MB each) + Y f32 [B,D,S] (16 MB)
    const size_t BSD = (size_t)BB * SS * DD;
    __bf16* Qb = (__bf16*)d_ws;
    __bf16* Kb = Qb + BSD;
    __bf16* Vb = Kb + BSD;
    __bf16* Cx = Vb + BSD;
    float*  Y  = (float*)(Cx + BSD);

    proj_kernel<<<1024, 128, 0, stream>>>(q, Wq, Qb);
    proj_kernel<<<1024, 128, 0, stream>>>(k, Wk, Kb);
    proj_kernel<<<1024, 128, 0, stream>>>(v, Wv, Vb);

    attn_kernel<<<2048, 128, 0, stream>>>(Qb, Kb, Vb, Cx);

    oproj_kernel<<<1024, 128, 0, stream>>>(Cx, Wo, Vb, Y);

    gn_kernel<<<BB * NG, 256, 0, stream>>>(Y, gamma, beta, (float*)d_out);
}
